// SVACrossAttentionLayer_43078521979058
// MI455X (gfx1250) — matrix operand order repaired and checked by reading, not compiled, not run on hardware
//
#include <hip/hip_runtime.h>
#include <hip/hip_bf16.h>

#define HID 1024
#define HEADS 16
#define HEAD_DIM 64
#define BB 2
#define NQ 1024
#define NK 4096
#define LN_EPS 1e-5f

typedef __attribute__((ext_vector_type(16))) __bf16        v16bf;
typedef __attribute__((ext_vector_type(8)))  float         v8f;
typedef __attribute__((ext_vector_type(4)))  unsigned int  u32x4;
typedef __attribute__((ext_vector_type(8)))  int           i32x8;
typedef __attribute__((ext_vector_type(4)))  int           i32x4;
typedef __attribute__((ext_vector_type(4)))  float         f32x4;

union Frag16 { u32x4 q[2]; v16bf v; };

#if __has_builtin(__builtin_amdgcn_global_load_async_to_lds_b128)
#define HAVE_ASYNC_LDS 1
#else
#define HAVE_ASYNC_LDS 0
#endif
#if __has_builtin(__builtin_amdgcn_tensor_load_to_lds)
#define HAVE_TDM 1
#else
#define HAVE_TDM 0
#endif

__device__ __forceinline__ unsigned short f2bf(float f) {
  unsigned int u = __float_as_uint(f);
  u += 0x7FFFu + ((u >> 16) & 1u);  // round-to-nearest-even
  return (unsigned short)(u >> 16);
}

// async global->LDS 16B copy (per-lane), fallback: plain load+store
typedef __attribute__((address_space(1))) i32x4 gvec128_t;  // global int4
typedef __attribute__((address_space(3))) i32x4 lvec128_t;  // LDS int4

__device__ __forceinline__ void copy16_g2l(const void* g, void* l) {
#if HAVE_ASYNC_LDS
  __builtin_amdgcn_global_load_async_to_lds_b128((gvec128_t*)g, (lvec128_t*)l, 0, 0);
#else
  *(u32x4*)l = *(const u32x4*)g;
#endif
}
__device__ __forceinline__ void copy16_wait() {
#if HAVE_ASYNC_LDS
#if __has_builtin(__builtin_amdgcn_s_wait_asynccnt)
  __builtin_amdgcn_s_wait_asynccnt(0);
#else
  asm volatile("s_wait_asynccnt 0x0" ::: "memory");
#endif
#endif
}

#if HAVE_TDM
// 2-D TDM tile load: tile0 x tile1 elements of 2 bytes, row stride stride0.
__device__ __forceinline__ void tdm_load_2d_bf16(const void* gaddr, unsigned lds_off,
                                                 unsigned tdim0, unsigned tdim1,
                                                 unsigned stride0, unsigned tile0,
                                                 unsigned tile1) {
  const unsigned long long ga = (unsigned long long)(uintptr_t)gaddr;
  u32x4 g0;
  g0.x = 1u;                                   // count=1, user descriptor
  g0.y = lds_off;                              // LDS byte address
  g0.z = (unsigned)(ga & 0xFFFFFFFFu);         // global_addr[31:0]
  g0.w = (unsigned)((ga >> 32) & 0x1FFFFFFu) | (2u << 30);  // addr[56:32], type=2
  i32x8 g1;
  g1[0] = (int)(1u << 16);                     // workgroup_mask=0, data_size=1 (2B)
  g1[1] = (int)(tdim0 << 16);                  // tensor_dim0[15:0] -> bits 63:48
  g1[2] = (int)((tdim0 >> 16) | (tdim1 << 16));
  g1[3] = (int)((tdim1 >> 16) | (tile0 << 16));
  g1[4] = (int)(tile1 & 0xFFFFu);              // tile_dim1; tile_dim2=0
  g1[5] = (int)stride0;                        // tensor_dim0_stride (data_size units)
  g1[6] = 0;
  g1[7] = 0;
  i32x4 z4 = {0, 0, 0, 0};
#if defined(__clang_major__) && __clang_major__ >= 23
  i32x8 z8 = {0, 0, 0, 0, 0, 0, 0, 0};
  __builtin_amdgcn_tensor_load_to_lds(g0, g1, z4, z4, z8, 0);
#else
  __builtin_amdgcn_tensor_load_to_lds(g0, g1, z4, z4, 0);
#endif
}
#endif

// ---------------------------------------------------------------------------
// LayerNorm (1024-wide row) -> bf16.  One block (256 thr) per row.
// ---------------------------------------------------------------------------
__global__ __launch_bounds__(256) void ln_to_bf16_kernel(
    const float* __restrict__ x, const float* __restrict__ w,
    const float* __restrict__ bias, unsigned short* __restrict__ y) {
  const int row = blockIdx.x;
  const int t = threadIdx.x;
  const f32x4* xr = (const f32x4*)(x + (size_t)row * HID);
  f32x4 v = xr[t];
  __shared__ float rs[256], rs2[256];
  rs[t]  = v.x + v.y + v.z + v.w;
  rs2[t] = v.x * v.x + v.y * v.y + v.z * v.z + v.w * v.w;
  __syncthreads();
  for (int off = 128; off > 0; off >>= 1) {
    if (t < off) { rs[t] += rs[t + off]; rs2[t] += rs2[t + off]; }
    __syncthreads();
  }
  const float mean = rs[0] * (1.0f / HID);
  const float var  = rs2[0] * (1.0f / HID) - mean * mean;
  const float rstd = rsqrtf(var + LN_EPS);
  const f32x4 wv = ((const f32x4*)w)[t];
  const f32x4 bv = ((const f32x4*)bias)[t];
  unsigned short* yr = y + (size_t)row * HID + t * 4;
  yr[0] = f2bf((v.x - mean) * rstd * wv.x + bv.x);
  yr[1] = f2bf((v.y - mean) * rstd * wv.y + bv.y);
  yr[2] = f2bf((v.z - mean) * rstd * wv.z + bv.z);
  yr[3] = f2bf((v.w - mean) * rstd * wv.w + bv.w);
}

// ---------------------------------------------------------------------------
// Residual + LayerNorm -> fp32 output.
// ---------------------------------------------------------------------------
__global__ __launch_bounds__(256) void residual_ln_kernel(
    const float* __restrict__ xq, const float* __restrict__ xp,
    const float* __restrict__ w, const float* __restrict__ bias,
    float* __restrict__ y) {
  const int row = blockIdx.x;
  const int t = threadIdx.x;
  const f32x4 a = ((const f32x4*)(xq + (size_t)row * HID))[t];
  const f32x4 b = ((const f32x4*)(xp + (size_t)row * HID))[t];
  f32x4 v; v.x = a.x + b.x; v.y = a.y + b.y; v.z = a.z + b.z; v.w = a.w + b.w;
  __shared__ float rs[256], rs2[256];
  rs[t]  = v.x + v.y + v.z + v.w;
  rs2[t] = v.x * v.x + v.y * v.y + v.z * v.z + v.w * v.w;
  __syncthreads();
  for (int off = 128; off > 0; off >>= 1) {
    if (t < off) { rs[t] += rs[t + off]; rs2[t] += rs2[t + off]; }
    __syncthreads();
  }
  const float mean = rs[0] * (1.0f / HID);
  const float var  = rs2[0] * (1.0f / HID) - mean * mean;
  const float rstd = rsqrtf(var + LN_EPS);
  const f32x4 wv = ((const f32x4*)w)[t];
  const f32x4 bv = ((const f32x4*)bias)[t];
  f32x4 o;
  o.x = (v.x - mean) * rstd * wv.x + bv.x;
  o.y = (v.y - mean) * rstd * wv.y + bv.y;
  o.z = (v.z - mean) * rstd * wv.z + bv.z;
  o.w = (v.w - mean) * rstd * wv.w + bv.w;
  ((f32x4*)(y + (size_t)row * HID))[t] = o;
}

// ---------------------------------------------------------------------------
// Wt[k][n] = bf16(scl * W[n][k]) — 32x32 LDS tile transpose + cast (+ folded
// score scale for Wq).
// ---------------------------------------------------------------------------
__global__ __launch_bounds__(256) void wt_bf16_kernel(
    const float* __restrict__ W, unsigned short* __restrict__ Wt, float scl) {
  __shared__ float tile[32][33];
  const int bx = blockIdx.x;  // k tile
  const int by = blockIdx.y;  // n tile
  const int tx = threadIdx.x & 31, ty = threadIdx.x >> 5;
#pragma unroll
  for (int i = 0; i < 32; i += 8)
    tile[ty + i][tx] = W[(size_t)(by * 32 + ty + i) * HID + bx * 32 + tx];
  __syncthreads();
#pragma unroll
  for (int i = 0; i < 32; i += 8)
    Wt[(size_t)(bx * 32 + ty + i) * HID + by * 32 + tx] =
        f2bf(scl * tile[tx][ty + i]);
}

// ---------------------------------------------------------------------------
// C[M,1024] = A[M,1024] * Bt[1024,1024]  (Bt[k][n] = W[n][k], i.e. x @ W^T)
// Block tile 128x64, 8 waves (4x2), wave tile 32x32 (2x2 WMMA accumulators),
// K-step 32.  A tile staged via TDM (Tensor Data Mover) when available.
// ---------------------------------------------------------------------------
template <bool OUT_BF16>
__global__ __launch_bounds__(256) void gemm_bf16_wmma(
    const unsigned short* __restrict__ A, const unsigned short* __restrict__ Bt,
    void* __restrict__ Cv, int M) {
  __shared__ alignas(16) unsigned short As[128 * 32];  // [m][k]
  __shared__ alignas(16) unsigned short Bs[64 * 32];   // [n][k] (restaged transposed)
  const int n0 = blockIdx.x * 64;
  const int m0 = blockIdx.y * 128;
  const int t = threadIdx.x;
  const int wave = t >> 5, lane = t & 31;
  const int wm = wave >> 1, wn = wave & 1;
  const bool lo = lane < 16;
  const int ln = lane & 15;

  v8f c[2][2] = {};

  const int arow = t >> 1, ahalf = t & 1;     // A staging: 128 rows x 2 halves
  const int bk = t & 31, bn0 = (t >> 5) * 8;  // B staging: 32 k x 8-n chunks

  for (int k0 = 0; k0 < HID; k0 += 32) {
    __syncthreads();
#if HAVE_TDM
    if (t < 32) {  // wave 0 issues the DMA; TDM ignores EXEC
      tdm_load_2d_bf16(A + (size_t)m0 * HID + k0,
                       (unsigned)(uintptr_t)(__attribute__((address_space(3))) void*)As,
                       /*tdim0=*/HID, /*tdim1=*/(unsigned)M, /*stride0=*/HID,
                       /*tile0=*/32, /*tile1=*/128);
      __builtin_amdgcn_s_wait_tensorcnt(0);
    }
    (void)arow; (void)ahalf;
#else
    {  // A tile -> LDS (natural layout)
      const u32x4* src = (const u32x4*)(A + (size_t)(m0 + arow) * HID + k0 + ahalf * 16);
      u32x4* dst = (u32x4*)(As + arow * 32 + ahalf * 16);
      dst[0] = src[0]; dst[1] = src[1];
    }
#endif
    {  // B tile -> LDS, transposed to [n][k]
      const u32x4 v = *(const u32x4*)(Bt + (size_t)(k0 + bk) * HID + n0 + bn0);
      const unsigned short* hs = (const unsigned short*)&v;
#pragma unroll
      for (int i = 0; i < 8; ++i) Bs[(bn0 + i) * 32 + bk] = hs[i];
    }
    if (k0 + 32 < HID) {  // gfx1250 global_prefetch_b8
      __builtin_prefetch(Bt + (size_t)(k0 + 32 + bk) * HID + n0 + bn0, 0, 1);
    }
    __syncthreads();

    Frag16 af[2], bf[2];
#pragma unroll
    for (int mi = 0; mi < 2; ++mi) {  // A 16x32 frag: lo lanes K0-7,16-23; hi K8-15,24-31
      const u32x4* ar = (const u32x4*)(As + (wm * 32 + mi * 16 + ln) * 32);
      af[mi].q[0] = ar[lo ? 0 : 1];
      af[mi].q[1] = ar[lo ? 2 : 3];
    }
#pragma unroll
    for (int ni = 0; ni < 2; ++ni) {  // B 32x16 frag: lane=n; lo K0-7,16-23; hi K8-15,24-31
      const u32x4* br = (const u32x4*)(Bs + (wn * 32 + ni * 16 + ln) * 32);
      bf[ni].q[0] = br[lo ? 0 : 1];
      bf[ni].q[1] = br[lo ? 2 : 3];
    }
#pragma unroll
    for (int mi = 0; mi < 2; ++mi)
#pragma unroll
      for (int ni = 0; ni < 2; ++ni)
        c[mi][ni] = __builtin_amdgcn_wmma_f32_16x16x32_bf16(
            false, af[mi].v, false, bf[ni].v, (short)0, c[mi][ni], false, false);
  }

#pragma unroll
  for (int mi = 0; mi < 2; ++mi)
#pragma unroll
    for (int ni = 0; ni < 2; ++ni)
#pragma unroll
      for (int r = 0; r < 8; ++r) {
        const int row = m0 + wm * 32 + mi * 16 + r + (lo ? 0 : 8);
        const int col = n0 + wn * 32 + ni * 16 + ln;
        if (OUT_BF16)
          ((unsigned short*)Cv)[(size_t)row * HID + col] = f2bf(c[mi][ni][r]);
        else
          ((float*)Cv)[(size_t)row * HID + col] = c[mi][ni][r];
      }
  (void)M;
}

// ---------------------------------------------------------------------------
// Flash attention.  Block = 128 thr = 4 waves sharing (b,h); each wave owns a
// 16-query tile.  32-key steps; K staged via async global->LDS, V staged with
// an in-LDS transpose; online softmax; P->A-layout via per-wave LDS round trip.
// Q arrives pre-scaled by 1/sqrt(head_dim) (folded into Wq).
// ---------------------------------------------------------------------------
__global__ __launch_bounds__(128) void attn_fmha_kernel(
    const unsigned short* __restrict__ Qp, const unsigned short* __restrict__ Kp,
    const unsigned short* __restrict__ Vp, const int* __restrict__ qtype,
    const int* __restrict__ ktype, const unsigned char* __restrict__ pad,
    const float* __restrict__ tbias, unsigned short* __restrict__ Out) {
  __shared__ alignas(16) unsigned short Ks[32 * 64];     // [j][d]
  __shared__ alignas(16) unsigned short Vs[64 * 32];     // [d][j]
  __shared__ alignas(16) unsigned short Ps[4][16 * 32];  // per-wave P tile

  const int t = threadIdx.x;
  const int wave = t >> 5, lane = t & 31;
  const bool lo = lane < 16;
  const int ln = lane & 15;
  const int h = blockIdx.y, b = blockIdx.z;
  const int q0 = blockIdx.x * 64 + wave * 16;

  // per-row typed-bias row (3 values) — constant indices inside the hot loop
  float brow[8][3];
#pragma unroll
  for (int r = 0; r < 8; ++r) {
    const int qv = qtype[q0 + r + (lo ? 0 : 8)];
#pragma unroll
    for (int i = 0; i < 3; ++i) brow[r][i] = tbias[qv * 3 + i];
  }

  Frag16 aq[2];  // Q 16x32 fragments for d-chunks [0,32) and [32,64)
  {
    const u32x4* qr = (const u32x4*)(Qp + (size_t)(b * NQ + q0 + ln) * HID + h * HEAD_DIM);
#pragma unroll
    for (int c = 0; c < 2; ++c) {
      aq[c].q[0] = qr[c * 4 + (lo ? 0 : 1)];
      aq[c].q[1] = qr[c * 4 + (lo ? 2 : 3)];
    }
  }

  v8f acc[4] = {};
  float m_i[8], l_i[8];
#pragma unroll
  for (int r = 0; r < 8; ++r) { m_i[r] = -1e30f; l_i[r] = 0.0f; }

  const int sj = t >> 2;   // staging key row 0..31
  const int sseg = t & 3;  // staging d segment of 16

  for (int kv0 = 0; kv0 < NK; kv0 += 32) {
    __syncthreads();
    {  // K tile [j][d] via async global->LDS (2 x b128 per lane)
      const unsigned short* src =
          Kp + (size_t)(b * NK + kv0 + sj) * HID + h * HEAD_DIM + sseg * 16;
      unsigned short* dst = Ks + sj * 64 + sseg * 16;
      copy16_g2l(src, dst);
      copy16_g2l(src + 8, dst + 8);
    }
    {  // V tile transposed [d][j]
      const u32x4 v0 = ((const u32x4*)(Vp + (size_t)(b * NK + kv0 + sj) * HID + h * HEAD_DIM + sseg * 16))[0];
      const u32x4 v1 = ((const u32x4*)(Vp + (size_t)(b * NK + kv0 + sj) * HID + h * HEAD_DIM + sseg * 16))[1];
      const unsigned short* h0 = (const unsigned short*)&v0;
      const unsigned short* h1 = (const unsigned short*)&v1;
#pragma unroll
      for (int e = 0; e < 8; ++e) {
        Vs[(sseg * 16 + e) * 32 + sj]     = h0[e];
        Vs[(sseg * 16 + 8 + e) * 32 + sj] = h1[e];
      }
    }
    copy16_wait();
    __syncthreads();

    // scores: two 16-key sub-tiles, K-dim = 64 as two chained WMMAs
    v8f s[2];
#pragma unroll
    for (int nt = 0; nt < 2; ++nt) {
      v8f sc = {};
#pragma unroll
      for (int c = 0; c < 2; ++c) {
        Frag16 kb;
        const u32x4* kr = (const u32x4*)(Ks + (nt * 16 + ln) * 64 + c * 32 + (lo ? 0 : 8));
        kb.q[0] = kr[0]; kb.q[1] = kr[2];
        sc = __builtin_amdgcn_wmma_f32_16x16x32_bf16(false, aq[c].v, false, kb.v,
                                                     (short)0, sc, false, false);
      }
      s[nt] = sc;
    }

    // typed bias (2 shared compares, constant-index selects) + padding mask
    float rmax[8];
#pragma unroll
    for (int r = 0; r < 8; ++r) rmax[r] = -1e30f;
#pragma unroll
    for (int nt = 0; nt < 2; ++nt) {
      const int j = kv0 + nt * 16 + ln;
      const int kt = ktype[j];
      const bool k1 = (kt == 1), k2 = (kt == 2);
      const float padd = pad[b * NK + j] ? 0.0f : -1e30f;
#pragma unroll
      for (int r = 0; r < 8; ++r) {
        const float bv = k1 ? brow[r][1] : (k2 ? brow[r][2] : brow[r][0]);
        const float val = s[nt][r] + bv + padd;
        s[nt][r] = val;
        rmax[r] = fmaxf(rmax[r], val);
      }
    }
    // online softmax (reductions stay inside each 16-lane C-layout group)
#pragma unroll
    for (int r = 0; r < 8; ++r) {
      float v = rmax[r];
#pragma unroll
      for (int m = 1; m < 16; m <<= 1) v = fmaxf(v, __shfl_xor(v, m, 32));
      const float mn = fmaxf(m_i[r], v);
      const float alpha = __expf(m_i[r] - mn);
      m_i[r] = mn;
      float psum = 0.0f;
#pragma unroll
      for (int nt = 0; nt < 2; ++nt) {
        const float p = __expf(s[nt][r] - mn);
        s[nt][r] = p;
        psum += p;
      }
#pragma unroll
      for (int m = 1; m < 16; m <<= 1) psum += __shfl_xor(psum, m, 32);
      l_i[r] = l_i[r] * alpha + psum;
#pragma unroll
      for (int dt = 0; dt < 4; ++dt) acc[dt][r] *= alpha;
    }

    // P (C-layout) -> per-wave LDS [M][j] -> A-layout 16x32 fragment
#pragma unroll
    for (int nt = 0; nt < 2; ++nt)
#pragma unroll
      for (int r = 0; r < 8; ++r)
        Ps[wave][(r + (lo ? 0 : 8)) * 32 + nt * 16 + ln] = f2bf(s[nt][r]);
    asm volatile("" ::: "memory");  // keep LDS write->read ordered (same wave)
    Frag16 pa;
    {
      const u32x4* pr = (const u32x4*)(Ps[wave] + ln * 32);
      pa.q[0] = pr[lo ? 0 : 1];
      pa.q[1] = pr[lo ? 2 : 3];
    }
    // P x V: 4 output d-tiles of 16
#pragma unroll
    for (int dt = 0; dt < 4; ++dt) {
      Frag16 vb;
      const u32x4* vr = (const u32x4*)(Vs + (dt * 16 + ln) * 32 + (lo ? 0 : 8));
      vb.q[0] = vr[0]; vb.q[1] = vr[2];
      acc[dt] = __builtin_amdgcn_wmma_f32_16x16x32_bf16(false, pa.v, false, vb.v,
                                                        (short)0, acc[dt], false, false);
    }
  }

#pragma unroll
  for (int r = 0; r < 8; ++r) {
    const float rl = 1.0f / fmaxf(l_i[r], 1e-20f);
    const int row = b * NQ + q0 + r + (lo ? 0 : 8);
#pragma unroll
    for (int dt = 0; dt < 4; ++dt)
      Out[(size_t)row * HID + h * HEAD_DIM + dt * 16 + ln] = f2bf(acc[dt][r] * rl);
  }
}

// ---------------------------------------------------------------------------
extern "C" void kernel_launch(void* const* d_in, const int* in_sizes, int n_in,
                              void* d_out, int out_size, void* d_ws, size_t ws_size,
                              hipStream_t stream) {
  const float* queries = (const float*)d_in[0];
  const float* kv      = (const float*)d_in[1];
  const int*   qtype   = (const int*)d_in[2];
  const int*   ktype   = (const int*)d_in[3];
  const unsigned char* pad = (const unsigned char*)d_in[4];  // bool mask
  const float* Wq = (const float*)d_in[5];
  const float* Wk = (const float*)d_in[6];
  const float* Wv = (const float*)d_in[7];
  const float* Wo = (const float*)d_in[8];
  const float* qn_w  = (const float*)d_in[9];
  const float* qn_b  = (const float*)d_in[10];
  const float* kvn_w = (const float*)d_in[11];
  const float* kvn_b = (const float*)d_in[12];
  const float* on_w  = (const float*)d_in[13];
  const float* on_b  = (const float*)d_in[14];
  const float* tbias = (const float*)d_in[15];
  float* out = (float*)d_out;

  char* ws = (char*)d_ws;
  const size_t MB = 1024 * 1024;
  unsigned short* qn    = (unsigned short*)(ws + 0);        //  4 MB
  unsigned short* kvn   = (unsigned short*)(ws + 4 * MB);   // 16 MB
  unsigned short* WqT   = (unsigned short*)(ws + 20 * MB);  //  2 MB
  unsigned short* WkT   = (unsigned short*)(ws + 22 * MB);
  unsigned short* WvT   = (unsigned short*)(ws + 24 * MB);
  unsigned short* WoT   = (unsigned short*)(ws + 26 * MB);
  unsigned short* qp    = (unsigned short*)(ws + 28 * MB);  //  4 MB
  unsigned short* kp    = (unsigned short*)(ws + 32 * MB);  // 16 MB
  unsigned short* vp    = (unsigned short*)(ws + 48 * MB);  // 16 MB
  unsigned short* ao    = (unsigned short*)(ws + 64 * MB);  //  4 MB
  float*          oproj = (float*)(ws + 68 * MB);           //  8 MB

  ln_to_bf16_kernel<<<BB * NQ, 256, 0, stream>>>(queries, qn_w, qn_b, qn);
  ln_to_bf16_kernel<<<BB * NK, 256, 0, stream>>>(kv, kvn_w, kvn_b, kvn);

  dim3 tg(32, 32);
  wt_bf16_kernel<<<tg, 256, 0, stream>>>(Wq, WqT, 0.125f);  // fold 1/sqrt(hd)
  wt_bf16_kernel<<<tg, 256, 0, stream>>>(Wk, WkT, 1.0f);
  wt_bf16_kernel<<<tg, 256, 0, stream>>>(Wv, WvT, 1.0f);
  wt_bf16_kernel<<<tg, 256, 0, stream>>>(Wo, WoT, 1.0f);

  gemm_bf16_wmma<true><<<dim3(16, 16), 256, 0, stream>>>(qn,  WqT, qp, BB * NQ);
  gemm_bf16_wmma<true><<<dim3(16, 64), 256, 0, stream>>>(kvn, WkT, kp, BB * NK);
  gemm_bf16_wmma<true><<<dim3(16, 64), 256, 0, stream>>>(kvn, WvT, vp, BB * NK);

  attn_fmha_kernel<<<dim3(16, 16, 2), 128, 0, stream>>>(qp, kp, vp, qtype, ktype,
                                                        pad, tbias, ao);

  gemm_bf16_wmma<false><<<dim3(16, 16), 256, 0, stream>>>(ao, WoT, oproj, BB * NQ);
  residual_ln_kernel<<<BB * NQ, 256, 0, stream>>>(queries, oproj, on_w, on_b, out);
}
